// AFNPBlock_256_74234214744675
// MI455X (gfx1250) — matrix operand order repaired and checked by reading, not compiled, not run on hardware
//
#include <hip/hip_runtime.h>
#include <hip/hip_bf16.h>

typedef _Float16 v16h __attribute__((ext_vector_type(16)));
typedef _Float16 v8h  __attribute__((ext_vector_type(8)));
typedef float    v8f  __attribute__((ext_vector_type(8)));

#define B_    4
#define C_    128
#define H_    256
#define W_    256
#define N_    (H_ * W_)
#define KTOT  396
#define KPAD  416
#define SCALEF 0.044194173824159216f   // 512^-0.5 (literal from source)

// ---------------------------------------------------------------------------
// Kernel 1: adaptive average pooling (all 7 scales) -> Key buffer + p12/p6
// One block per (b,c) plane. Separable: row-pool into LDS, then column-pool.
// ---------------------------------------------------------------------------
__global__ __launch_bounds__(256) void pool_kernel(const float* __restrict__ x5,
                                                   float* __restrict__ keybuf,
                                                   float* __restrict__ out) {
  __shared__ float rowPool[256 * 55];   // 54 row-pooled cols per row, +1 pad
  const int plane = blockIdx.x;         // b*C + c
  const float* img = x5 + (size_t)plane * N_;
  const int y = threadIdx.x;            // 256 threads = 256 rows

  // scale order matches Key concat: p3, p1, p9, p7, p16, then p12, p6 extras
  const int sz[7]   = {3, 1, 9, 7, 16, 12, 6};
  const int rOff[7] = {0, 3, 4, 13, 20, 36, 48};      // row-pool col offsets (sum 54)
  const int oOff[7] = {0, 9, 10, 91, 140, 396, 540};  // flat pooled offsets (sum 576)

  // ---- phase 1: pool along W for this thread's row ----
  {
    const float* row = img + (size_t)y * W_;
    __builtin_prefetch(row, 0, 3);
    #pragma unroll
    for (int si = 0; si < 7; ++si) {
      const int s = sz[si];
      for (int k = 0; k < s; ++k) {
        const int ws = (k * W_) / s;
        const int we = ((k + 1) * W_ + s - 1) / s;  // ceil
        float acc = 0.f;
        for (int xx = ws; xx < we; ++xx) acc += row[xx];
        rowPool[y * 55 + rOff[si] + k] = acc;
      }
    }
  }
  __syncthreads();

  // ---- phase 2: pool along H, normalize, scatter to destinations ----
  for (int e = threadIdx.x; e < 576; e += 256) {
    int si = 0;
    if (e >= 9)   si = 1;
    if (e >= 10)  si = 2;
    if (e >= 91)  si = 3;
    if (e >= 140) si = 4;
    if (e >= 396) si = 5;
    if (e >= 540) si = 6;
    const int s   = sz[si];
    const int loc = e - oOff[si];
    const int oy  = loc / s;
    const int ox  = loc - oy * s;
    const int ys = (oy * H_) / s, ye = ((oy + 1) * H_ + s - 1) / s;
    const int xs = (ox * W_) / s, xe = ((ox + 1) * W_ + s - 1) / s;
    float acc = 0.f;
    for (int yy = ys; yy < ye; ++yy) acc += rowPool[yy * 55 + rOff[si] + ox];
    const float val = acc / (float)((ye - ys) * (xe - xs));
    if (si <= 4) {
      // Key concat order [p3,p1,p9,p7,p16] -> offsets identical to oOff
      keybuf[(size_t)plane * KTOT + oOff[si] + loc] = val;
    } else if (si == 5) {  // p12 -> second tuple output
      out[(size_t)B_ * C_ * N_ + (size_t)plane * 144 + loc] = val;
    } else {               // p6 -> third tuple output
      out[(size_t)B_ * C_ * N_ + (size_t)B_ * C_ * 144 + (size_t)plane * 36 + loc] = val;
    }
  }
}

// ---------------------------------------------------------------------------
// Kernel 2: fused attention + 1x1 conv + bias + residual (WMMA f16->f32)
// One block per (batch, 64-row query tile). 4 waves, each owns 16 rows.
// Key kept in BOTH layouts in LDS so every WMMA B-fragment is two aligned
// 16-byte ds_load_b128 (no strided ds_load_u16 gathers).
// ---------------------------------------------------------------------------
#define TM    64                          // query rows per block
#define QSTR  136                         // halves; 272B rows, 16B-aligned
#define TSTR  136                         // keyT row stride (halves)
#define OSTR  65                          // outT floats stride (odd -> bank spread)
#define OFF_KEY 0
#define SZ_KEY  (128 * KPAD * 2)          // 106496  keyL [c][KPAD]
#define OFF_KT  (OFF_KEY + SZ_KEY)        // 106496
#define SZ_KT   (KPAD * TSTR * 2)         // 113152  keyT [k][TSTR]
#define OFF_Q   (OFF_KT + SZ_KT)          // 219648
#define SZ_Q    (TM * QSTR * 2)           // 17408   qL [n][QSTR] (later agg)
#define OFF_SP  (OFF_Q + SZ_Q)            // 237056
#define SZ_SP   (TM * KPAD * 2)           // 53248   spL; outT needs 128*65*4=33280
#define OFF_W   (OFF_SP + SZ_SP)          // 290304
#define SZ_W    (128 * 128 * 2)           // 32768   wL [o][c]
#define OFF_R   (OFF_W + SZ_W)            // 323072
#define SMEM_SZ (OFF_R + TM * 4)          // 323328 bytes (<= 320KB WGP LDS)

__global__ __launch_bounds__(128) void attn_kernel(const float* __restrict__ x5,
                                                   const float* __restrict__ keybuf,
                                                   const float* __restrict__ Wout,
                                                   const float* __restrict__ bout,
                                                   float* __restrict__ out) {
  extern __shared__ char smem[];
  _Float16* keyL = (_Float16*)(smem + OFF_KEY);  // [c][KPAD]  Key (contig in k)
  _Float16* keyT = (_Float16*)(smem + OFF_KT);   // [k][TSTR]  Key^T (contig in c)
  _Float16* qL   = (_Float16*)(smem + OFF_Q);    // [n][QSTR]  Q tile, later agg
  _Float16* spL  = (_Float16*)(smem + OFF_SP);   // [n][KPAD]  S then P
  float*    outT = (float*)(smem + OFF_SP);      // [o][OSTR]  aliased after P dead
  _Float16* wL   = (_Float16*)(smem + OFF_W);    // [o][c]     W_out (natural)
  float*    rinv = (float*)(smem + OFF_R);       // [TM] softmax 1/rowsum

  const int b    = blockIdx.x >> 10;             // 1024 tiles per batch
  const int n0   = (blockIdx.x & 1023) << 6;
  const int tid  = threadIdx.x;
  const int lane = tid & 31;
  const int w    = tid >> 5;       // wave id 0..3; owns query rows w*16..w*16+15
  const int lhi  = lane >> 4;      // lane half
  const int l15  = lane & 15;
  // f16 K bases of the ISA 16-bit A 16x32 layout (unused; fragments use bko):
  const int a0  = lhi ? 8 : 0;     // halves 0..7  -> K = a0 + h
  const int a1  = lhi ? 24 : 16;   // halves 8..15 -> K = a1 + (h-8)
  // f16 K base for both A and B fragments (contiguous order): halves 0..15 -> K = bko + h
  const int bko = lhi ? 16 : 0;

  // ---- cooperative LDS staging (coalesced global reads, dual Key layout) ----
  for (int e = tid; e < 128 * KPAD; e += 128) {
    const int c = e / KPAD, k = e - c * KPAD;
    const float v = (k < KTOT) ? keybuf[(size_t)(b * C_ + c) * KTOT + k] : 0.f;
    const _Float16 hv = (_Float16)v;
    keyL[c * KPAD + k] = hv;
    keyT[k * TSTR + c] = hv;
  }
  for (int e = tid; e < TM * 128; e += 128) {
    const int c = e >> 6, n = e & (TM - 1);
    qL[n * QSTR + c] = (_Float16)x5[(size_t)(b * C_ + c) * N_ + n0 + n];
  }
  for (int e = tid; e < 128 * 128; e += 128) wL[e] = (_Float16)Wout[e];
  __syncthreads();

  const int qrow = w * 16 + l15;   // A-fragment row for this lane

  // ---- phase A: S = SCALE * Q @ Key  (25 key tiles x 4 channel chunks) ----
  for (int kt = 0; kt < 25; ++kt) {
    v8f s = {0.f, 0.f, 0.f, 0.f, 0.f, 0.f, 0.f, 0.f};
    #pragma unroll
    for (int cc = 0; cc < 4; ++cc) {
      union { v16h v; v8h h[2]; } A;
      A.h[0] = *(const v8h*)(qL + qrow * QSTR + cc * 32 + bko);
      A.h[1] = *(const v8h*)(qL + qrow * QSTR + cc * 32 + bko + 8);
      // B[K=c][N=k] = keyT[k][c]: 16 contiguous halves in c
      union { v16h v; v8h h[2]; } Bv;
      const int krow = (kt * 16 + l15) * TSTR + cc * 32 + bko;
      Bv.h[0] = *(const v8h*)(keyT + krow);
      Bv.h[1] = *(const v8h*)(keyT + krow + 8);
      s = __builtin_amdgcn_wmma_f32_16x16x32_f16(false, A.v, false, Bv.v,
                                                 (short)0, s, false, false);
    }
    #pragma unroll
    for (int v = 0; v < 8; ++v) {        // C/D layout: row = v + 8*lhi, col = l15
      const int r = w * 16 + v + 8 * lhi;
      spL[r * KPAD + kt * 16 + l15] = (_Float16)(s[v] * SCALEF);
    }
  }
  __syncthreads();

  // ---- phase B: rowwise softmax on this wave's 16 rows (pad cols -> P=0) ----
  for (int r0 = 0; r0 < 16; ++r0) {
    const int r = w * 16 + r0;
    float m = -3.0e38f;
    for (int k = lane; k < KTOT; k += 32) m = fmaxf(m, (float)spL[r * KPAD + k]);
    #pragma unroll
    for (int off = 16; off > 0; off >>= 1) m = fmaxf(m, __shfl_xor(m, off, 32));
    float sum = 0.f;
    for (int k = lane; k < KPAD; k += 32) {
      float p = 0.f;
      if (k < KTOT) { p = __expf((float)spL[r * KPAD + k] - m); sum += p; }
      spL[r * KPAD + k] = (_Float16)p;   // pad columns zeroed
    }
    #pragma unroll
    for (int off = 16; off > 0; off >>= 1) sum += __shfl_xor(sum, off, 32);
    if (lane == 0) rinv[r] = 1.f / sum;
  }
  __syncthreads();

  // ---- phase C: agg = P @ Key^T  (13 K-chunks of 32, 8 channel tiles) ----
  v8f acc[8];
  #pragma unroll
  for (int ct = 0; ct < 8; ++ct) acc[ct] = (v8f){0.f,0.f,0.f,0.f,0.f,0.f,0.f,0.f};
  for (int kc = 0; kc < 13; ++kc) {
    union { v16h v; v8h h[2]; } A;       // A = P rows (own wave's rows)
    A.h[0] = *(const v8h*)(spL + qrow * KPAD + kc * 32 + bko);
    A.h[1] = *(const v8h*)(spL + qrow * KPAD + kc * 32 + bko + 8);
    #pragma unroll
    for (int ct = 0; ct < 8; ++ct) {
      const int c = ct * 16 + l15;       // B[K=k][N=c] = keyL[c][k]: contig in k
      union { v16h v; v8h h[2]; } Bv;
      Bv.h[0] = *(const v8h*)(keyL + c * KPAD + kc * 32 + bko);
      Bv.h[1] = *(const v8h*)(keyL + c * KPAD + kc * 32 + bko + 8);
      acc[ct] = __builtin_amdgcn_wmma_f32_16x16x32_f16(false, A.v, false, Bv.v,
                                                       (short)0, acc[ct], false, false);
    }
  }
  {  // softmax normalization (deferred 1/rowsum)
    float rv[8];
    #pragma unroll
    for (int v = 0; v < 8; ++v) rv[v] = rinv[w * 16 + v + 8 * lhi];
    #pragma unroll
    for (int ct = 0; ct < 8; ++ct)
      #pragma unroll
      for (int v = 0; v < 8; ++v) acc[ct][v] *= rv[v];
  }

  // ---- phase D: out = agg @ W_out^T, transpose via LDS, +bias +residual ----
  __syncthreads();                       // all waves done reading Q and P regions
  #pragma unroll
  for (int ct = 0; ct < 8; ++ct)         // stage agg (f16) into the Q buffer
    #pragma unroll
    for (int v = 0; v < 8; ++v) {
      const int r = w * 16 + v + 8 * lhi;
      qL[r * QSTR + ct * 16 + l15] = (_Float16)acc[ct][v];
    }
  __syncthreads();
  for (int ot = 0; ot < 8; ++ot) {
    v8f o2 = {0.f, 0.f, 0.f, 0.f, 0.f, 0.f, 0.f, 0.f};
    #pragma unroll
    for (int cc = 0; cc < 4; ++cc) {
      union { v16h v; v8h h[2]; } A;
      A.h[0] = *(const v8h*)(qL + qrow * QSTR + cc * 32 + bko);
      A.h[1] = *(const v8h*)(qL + qrow * QSTR + cc * 32 + bko + 8);
      const int oo = ot * 16 + l15;      // B[K=c][N=o] = wL[o][c]: contig in c
      union { v16h v; v8h h[2]; } Bv;
      Bv.h[0] = *(const v8h*)(wL + oo * 128 + cc * 32 + bko);
      Bv.h[1] = *(const v8h*)(wL + oo * 128 + cc * 32 + bko + 8);
      o2 = __builtin_amdgcn_wmma_f32_16x16x32_f16(false, A.v, false, Bv.v,
                                                  (short)0, o2, false, false);
    }
    #pragma unroll
    for (int v = 0; v < 8; ++v)          // transpose into [o][n] (odd stride)
      outT[(ot * 16 + l15) * OSTR + (w * 16 + v + 8 * lhi)] = o2[v];
  }
  __syncthreads();
  for (int e = tid; e < 128 * TM; e += 128) {    // coalesced epilogue store
    const int o = e >> 6, n = e & (TM - 1);
    const size_t gi = (size_t)(b * C_ + o) * N_ + n0 + n;
    out[gi] = outT[o * OSTR + n] + bout[o] + x5[gi];
  }
}

// ---------------------------------------------------------------------------
extern "C" void kernel_launch(void* const* d_in, const int* in_sizes, int n_in,
                              void* d_out, int out_size, void* d_ws, size_t ws_size,
                              hipStream_t stream) {
  (void)in_sizes; (void)n_in; (void)out_size; (void)ws_size;
  const float* x5   = (const float*)d_in[0];
  const float* Wout = (const float*)d_in[1];
  const float* bout = (const float*)d_in[2];
  float* out    = (float*)d_out;
  float* keybuf = (float*)d_ws;          // B*C*396 floats (~0.8 MB)

  pool_kernel<<<B_ * C_, 256, 0, stream>>>(x5, keybuf, out);

  (void)hipFuncSetAttribute((const void*)attn_kernel,
                            hipFuncAttributeMaxDynamicSharedMemorySize, SMEM_SZ);
  attn_kernel<<<B_ * (N_ / TM), 128, SMEM_SZ, stream>>>(x5, keybuf, Wout, bout, out);
}
